// Router_55688545960289
// MI455X (gfx1250) — compile-verified, matrix operand order checked by reading, not run on hardware
//
#include <hip/hip_runtime.h>
#include <hip/hip_bf16.h>

// ---------------------------------------------------------------------------
// MoE router, fused:  h = relu(x@W1+b1); logits = h@W2+b2; softmax; top-2.
//   x  : [32768, 2048] f32   (8*4096 tokens)
//   W1 : [2048, 1024]  f32   -> converted once to bf16 column-major [1024][2048]
//   W2 : [1024, 16]    f32   -> converted once to bf16 column-major [16][1024]
// Outputs (concatenated in d_out): gates [N,2] f32 | indices [N,2] i32 |
//                                  probs [N,16] f32
// Matrix math: v_wmma_f32_16x16x32_bf16 (f32 accumulate), 4 independent
// accumulator chains per wave (2M x 2N register tile) to cover the bf16
// WMMA->WMMA hazard window without v_nop padding.
// ---------------------------------------------------------------------------

#define D_MODEL   2048
#define D_HID     1024
#define NE        16
#define TOPK      2
#define N_TOKENS  32768
#define M_TILE    32
#define A_STRIDE  (D_MODEL + 8)   // 2056 bf16 -> 4112B row stride, conflict-free b128
#define H_STRIDE  (D_HID + 8)     // 1032 bf16 -> 2064B row stride

typedef __attribute__((ext_vector_type(16))) __bf16 v16bf;
typedef __attribute__((ext_vector_type(8)))  __bf16 v8bf;
typedef __attribute__((ext_vector_type(8)))  float  v8f;
typedef __attribute__((ext_vector_type(4)))  float  v4f;

// --- prep: W1 [2048,1024] f32 row-major  ->  W1bf [1024][2048] bf16 (per-column rows)
__global__ __launch_bounds__(256) void cvt_w1(const float* __restrict__ W1,
                                              __bf16* __restrict__ W1bf) {
    int id = blockIdx.x * 256 + threadIdx.x;   // [0, 1024*2048)
    int n  = id >> 11;                         // output column of GEMM1
    int k  = id & (D_MODEL - 1);
    W1bf[id] = (__bf16)W1[(size_t)k * D_HID + n];
}

// --- prep: W2 [1024,16] f32 -> W2bf [16][1024] bf16
__global__ __launch_bounds__(256) void cvt_w2(const float* __restrict__ W2,
                                              __bf16* __restrict__ W2bf) {
    int id = blockIdx.x * 256 + threadIdx.x;   // [0, 16*1024)
    int e  = id >> 10;
    int k  = id & (D_HID - 1);
    W2bf[id] = (__bf16)W2[(size_t)k * NE + e];
}

__global__ __launch_bounds__(256) void router_fused(
    const float*  __restrict__ x,
    const float*  __restrict__ b1,
    const float*  __restrict__ b2,
    const __bf16* __restrict__ W1bf,
    const __bf16* __restrict__ W2bf,
    float* __restrict__ gates,
    int*   __restrict__ topidx,
    float* __restrict__ probs)
{
    __shared__ __bf16 As[M_TILE][A_STRIDE];   // 131584 B  staged x tile (bf16)
    __shared__ __bf16 Hs[M_TILE][H_STRIDE];   //  66048 B  hidden activations
    __shared__ float  Lg[M_TILE][NE];         //   2048 B  logits

    const int tid  = threadIdx.x;
    const int wave = tid >> 5;
    const int lane = tid & 31;
    const int m0   = blockIdx.x * M_TILE;

    // ---- Phase A: cooperative load x[32,2048] f32 -> bf16 LDS ---------------
    for (int i = tid; i < (M_TILE * D_MODEL) / 4; i += 256) {
        int row = i >> 9;                 // / (2048/4)
        int col = (i & 511) << 2;
        v4f f = *(const v4f*)(x + (size_t)(m0 + row) * D_MODEL + col);
        As[row][col + 0] = (__bf16)f.x;
        As[row][col + 1] = (__bf16)f.y;
        As[row][col + 2] = (__bf16)f.z;
        As[row][col + 3] = (__bf16)f.w;
    }
    __syncthreads();

    // WMMA 16-bit A-layout addressing (ISA 7.12.2):
    //   lane r = lane&15 holds row M=r; K-group ko = (lane>>4)*8,
    //   frag[0..7] = K[k0+ko .. +7], frag[8..15] = K[k0+16+ko .. +7]
    const int r  = lane & 15;
    const int ko = (lane >> 4) << 3;
    const int mro = (lane >> 4) << 3;     // C-layout row offset

    // ---- Phase B: GEMM1 + bias + ReLU --------------------------------------
    // Each wave: 32x32 register tile = 4 independent WMMA chains (2M x 2N).
    // 8 waves x 2 adjacent N-tiles per pass, 4 passes cover D_HID/16 = 64 tiles.
#pragma unroll 1
    for (int p = 0; p < 4; ++p) {
        const int n0 = (wave * 2 + p * 16) * 16;        // first of two N-tiles
        v8f acc00 = {}, acc01 = {}, acc10 = {}, acc11 = {};
        const __bf16* bcol0 = W1bf + (size_t)(n0 + r) * D_MODEL;       // L2-resident
        const __bf16* bcol1 = bcol0 + (size_t)16 * D_MODEL;
        for (int k0 = 0; k0 < D_MODEL; k0 += 32) {
            v16bf a0, a1, bb0, bb1;
            ((v8bf*)&a0)[0]  = *(const v8bf*)&As[r     ][k0 + ko];
            ((v8bf*)&a0)[1]  = *(const v8bf*)&As[r     ][k0 + 16 + ko];
            ((v8bf*)&a1)[0]  = *(const v8bf*)&As[r + 16][k0 + ko];
            ((v8bf*)&a1)[1]  = *(const v8bf*)&As[r + 16][k0 + 16 + ko];
            ((v8bf*)&bb0)[0] = *(const v8bf*)(bcol0 + k0 + ko);
            ((v8bf*)&bb0)[1] = *(const v8bf*)(bcol0 + k0 + 16 + ko);
            ((v8bf*)&bb1)[0] = *(const v8bf*)(bcol1 + k0 + ko);
            ((v8bf*)&bb1)[1] = *(const v8bf*)(bcol1 + k0 + 16 + ko);
            acc00 = __builtin_amdgcn_wmma_f32_16x16x32_bf16(
                        false, a0, false, bb0, (short)0, acc00, false, false);
            acc10 = __builtin_amdgcn_wmma_f32_16x16x32_bf16(
                        false, a1, false, bb0, (short)0, acc10, false, false);
            acc01 = __builtin_amdgcn_wmma_f32_16x16x32_bf16(
                        false, a0, false, bb1, (short)0, acc01, false, false);
            acc11 = __builtin_amdgcn_wmma_f32_16x16x32_bf16(
                        false, a1, false, bb1, (short)0, acc11, false, false);
        }
        // C-layout: element j -> row (j + 8*(lane>>4)), col (n0 + r)
        const float bias0 = b1[n0 + r];
        const float bias1 = b1[n0 + 16 + r];
#pragma unroll
        for (int j = 0; j < 8; ++j) {
            float h00 = acc00[j] + bias0; h00 = h00 > 0.f ? h00 : 0.f;
            float h10 = acc10[j] + bias0; h10 = h10 > 0.f ? h10 : 0.f;
            float h01 = acc01[j] + bias1; h01 = h01 > 0.f ? h01 : 0.f;
            float h11 = acc11[j] + bias1; h11 = h11 > 0.f ? h11 : 0.f;
            Hs[mro + j     ][n0 + r]      = (__bf16)h00;
            Hs[mro + j + 16][n0 + r]      = (__bf16)h10;
            Hs[mro + j     ][n0 + 16 + r] = (__bf16)h01;
            Hs[mro + j + 16][n0 + 16 + r] = (__bf16)h11;
        }
    }
    __syncthreads();

    // ---- Phase C: GEMM2 (N = 16 experts = one WMMA tile) -------------------
    if (wave < 2) {
        v8f acc = {};
        const int mrow0 = wave * 16;
        const __bf16* wcol = W2bf + (size_t)r * D_HID;  // expert column r
        for (int k0 = 0; k0 < D_HID; k0 += 32) {
            v16bf a, bb;
            ((v8bf*)&a)[0]  = *(const v8bf*)&Hs[mrow0 + r][k0 + ko];
            ((v8bf*)&a)[1]  = *(const v8bf*)&Hs[mrow0 + r][k0 + 16 + ko];
            ((v8bf*)&bb)[0] = *(const v8bf*)(wcol + k0 + ko);
            ((v8bf*)&bb)[1] = *(const v8bf*)(wcol + k0 + 16 + ko);
            acc = __builtin_amdgcn_wmma_f32_16x16x32_bf16(
                      false, a, false, bb, (short)0, acc, false, false);
        }
        const float bias = b2[r];
#pragma unroll
        for (int j = 0; j < 8; ++j)
            Lg[mrow0 + mro + j][r] = acc[j] + bias;
    }
    __syncthreads();

    // ---- Phase D: softmax over 16 experts, top-2, gate renorm --------------
    if (tid < M_TILE) {
        const int g = m0 + tid;
        float l[NE];
        float mx = -3.402823466e38f;
#pragma unroll
        for (int e = 0; e < NE; ++e) { l[e] = Lg[tid][e]; mx = l[e] > mx ? l[e] : mx; }
        float s = 0.f;
#pragma unroll
        for (int e = 0; e < NE; ++e) { l[e] = __expf(l[e] - mx); s += l[e]; }
        const float inv = 1.f / s;
        float p1 = -1.f, p2 = -1.f; int i1 = 0, i2 = 0;
#pragma unroll
        for (int e = 0; e < NE; ++e) {
            float p = l[e] * inv;
            probs[(size_t)g * NE + e] = p;
            if (p > p1)      { p2 = p1; i2 = i1; p1 = p; i1 = e; }
            else if (p > p2) { p2 = p;  i2 = e; }
        }
        const float sn = p1 + p2 + 1e-8f;
        gates[2 * g + 0]  = p1 / sn;
        gates[2 * g + 1]  = p2 / sn;
        topidx[2 * g + 0] = i1;
        topidx[2 * g + 1] = i2;
    }
}

extern "C" void kernel_launch(void* const* d_in, const int* in_sizes, int n_in,
                              void* d_out, int out_size, void* d_ws, size_t ws_size,
                              hipStream_t stream) {
    (void)in_sizes; (void)n_in; (void)out_size; (void)ws_size;

    const float* x  = (const float*)d_in[0];   // hidden_states [8,4096,2048]
    const float* W1 = (const float*)d_in[1];   // [2048,1024]
    const float* b1 = (const float*)d_in[2];   // [1024]
    const float* W2 = (const float*)d_in[3];   // [1024,16]
    const float* b2 = (const float*)d_in[4];   // [16]

    __bf16* W1bf = (__bf16*)d_ws;                       // 4 MiB
    __bf16* W2bf = W1bf + (size_t)D_HID * D_MODEL;      // 32 KiB

    float* gates = (float*)d_out;                       // [N,2] f32
    int*   tidx  = (int*)d_out + N_TOKENS * TOPK;       // [N,2] i32
    float* prob  = (float*)d_out + N_TOKENS * 2 * TOPK; // [N,16] f32

    cvt_w1<<<(D_HID * D_MODEL) / 256, 256, 0, stream>>>(W1, W1bf);
    cvt_w2<<<(NE * D_HID) / 256, 256, 0, stream>>>(W2, W2bf);
    router_fused<<<N_TOKENS / M_TILE, 256, 0, stream>>>(
        x, b1, b2, W1bf, W2bf, gates, tidx, prob);
}
